// DynamicMaskHead_26508538151624
// MI455X (gfx1250) — matrix operand order repaired and checked by reading, not compiled, not run on hardware
//
#include <hip/hip_runtime.h>
#include <hip/hip_bf16.h>

// ---------------------------------------------------------------------------
// DynamicMaskHead (CondInst-style dynamic conv head) for MI455X / gfx1250.
//
// Per instance (128): 3-layer 1x1-conv MLP over 9600 pixels.
//   L0: (128x66)*(66xP)+b0, relu   L1: (128x128)*(128xP)+b1, relu
//   L2: (1x128)*(128xP)+b2-2.19
// ~61 GFLOP vs ~28 MB traffic -> WMMA f16 path (v_wmma_f32_16x16x32_f16)
// with f32 accumulate; activations staged in LDS in [pixel][channel] layout
// so B-fragments are wide ds_load_b128 and C stores are packed ds_store_b128.
// ---------------------------------------------------------------------------

#define CH      128
#define INCH    64
#define CIN     66          // IN_CH + 2 coord channels
#define K0PAD   96          // layer-0 K padded to 3x32
#define IMG_W   160
#define NPIX    9600        // 60*160
#define TILE    128         // pixels per workgroup
#define NTILES  75          // 9600 / 128
#define NPAR    25217       // params per instance
#define OFF_W1  8448
#define OFF_W2  24832
#define OFF_B0  24960
#define OFF_B1  25088
#define OFF_B2  25216
#define MASK_SHIFT 2.19f
#define FPITCH  (K0PAD + 8)  // 104 f16 = 208 B (16B multiple, bank-staggered)
#define HPITCH  (CH + 8)     // 136 f16 = 272 B (16B multiple, bank-staggered)

typedef __attribute__((ext_vector_type(16))) _Float16 v16h;
typedef __attribute__((ext_vector_type(8)))  _Float16 v8h;
typedef __attribute__((ext_vector_type(4)))  _Float16 v4h;
typedef __attribute__((ext_vector_type(8)))  float    v8f;

// Concatenate two 16B-aligned LDS halves into one v16h B-fragment
// (lowers to two ds_load_b128 into adjacent VGPR quads).
__device__ __forceinline__ v16h ldsLoadFrag16(const _Float16* p) {
    v8h lo = *(const v8h*)p;
    v8h hi = *(const v8h*)(p + 8);
    return __builtin_shufflevector(lo, hi, 0, 1, 2, 3, 4, 5, 6, 7,
                                           8, 9, 10, 11, 12, 13, 14, 15);
}

__global__ __launch_bounds__(256)
void dyn_mask_head_kernel(const float* __restrict__ x,
                          const float* __restrict__ params,
                          const int*   __restrict__ num_ins,
                          int n_images,
                          float* __restrict__ out)
{
    // [pixel][channel] activation layout: B-fragments contiguous per lane.
    __shared__ __attribute__((aligned(16))) _Float16 sF[TILE][FPITCH]; // layer-0 B (coords+x)
    __shared__ __attribute__((aligned(16))) _Float16 sH[TILE][HPITCH]; // layer-0 out / layer-1 B
    __shared__ float sB0[CH], sB1[CH], sW2[CH];
    __shared__ float sOut[TILE];                 // fused layer-2 accumulator

    const int tile = blockIdx.x;                 // 0..74
    const int inst = blockIdx.y;                 // 0..127
    const int t    = threadIdx.x;                // 0..255
    const int lane = t & 31;
    const int wave = t >> 5;                     // 0..7 (wave32)
    const int px0  = tile * TILE;

    const float* pr = params + (size_t)inst * NPAR;

    // --- which source image feeds this instance (prefix sum of num_ins) ---
    int img = 0, csum = 0;
    for (int n = 0; n < n_images; ++n) {
        csum += num_ins[n];
        if (inst >= csum) img = n + 1;
    }
    const float* xi = x + (size_t)img * INCH * NPIX;

    // --- stage biases / w2 / output accumulator ---
    if (t < CH) {
        sB0[t] = pr[OFF_B0 + t];
        sB1[t] = pr[OFF_B1 + t];
        sW2[t] = pr[OFF_W2 + t];
    }
    if (t < TILE) sOut[t] = pr[OFF_B2] - MASK_SHIFT;

    // --- stage feature tile sF[p][k], 4 channels packed per ds_store_b64 ---
    // Reference quirk reproduced: BOTH coord channels divided by W.
    for (int idx = t; idx < TILE * (K0PAD / 4); idx += 256) {
        int p  = idx & (TILE - 1);
        int k  = (idx >> 7) * 4;
        int px = px0 + p;
        v4h pk;
#pragma unroll
        for (int j = 0; j < 4; ++j) {
            int kk = k + j;
            float v;
            if (kk == 0)        v = (float)(px % IMG_W) * (1.0f / IMG_W);
            else if (kk == 1)   v = (float)(px / IMG_W) * (1.0f / IMG_W);
            else if (kk < CIN)  v = xi[(size_t)(kk - 2) * NPIX + px];
            else                v = 0.0f;
            pk[j] = (_Float16)v;
        }
        *(v4h*)&sF[p][k] = pk;                  // 8B-aligned packed store
    }

    // --- per-wave weight A-fragments in registers (f32 -> f16) ---
    // A frags (16x32 f16, this kernel's contiguous k order): lane l holds row m = 16*wave + (l&15);
// element e maps to K = kb*32 + (l>>4)*16 + e.
    const int mrow  = wave * 16 + (lane & 15);
    const int khalf = (lane >> 4) * 8;

    v16h a0[3];
    {
        const float* w0r = pr + mrow * CIN;
        for (int kb = 0; kb < 3; ++kb) {
            v16h a;
#pragma unroll
            for (int e = 0; e < 16; ++e) {
                int K = kb * 32 + 2 * khalf + e;
                a[e] = (K < CIN) ? (_Float16)w0r[K] : (_Float16)0.0f;
            }
            a0[kb] = a;
        }
    }
    v16h a1[4];
    {
        const float* w1r = pr + OFF_W1 + mrow * CH;
        for (int kb = 0; kb < 4; ++kb) {
            v16h a;
#pragma unroll
            for (int e = 0; e < 16; ++e) {
                int K = kb * 32 + 2 * khalf + e;
                a[e] = (_Float16)w1r[K];
            }
            a1[kb] = a;
        }
    }

    __syncthreads();

    // C/D layout (16x16 f32): lane l -> col n = l&15; VGPR r -> row r + 8*(l>>4).
    const int col   = lane & 15;
    const int rbase = wave * 16 + 8 * (lane >> 4);
    const int kpart = (lane >> 4) * 16;          // lane-half K offset for B frags

    // ----------------- layer 0: h0 = relu(w0 * f + b0) -----------------
    for (int n = 0; n < 8; ++n) {
        const int p = n * 16 + col;
        v8f c;
#pragma unroll
        for (int r = 0; r < 8; ++r) c[r] = sB0[rbase + r];
#pragma unroll
        for (int kb = 0; kb < 3; ++kb) {
            // B layout (32x16 f16): lane l -> col l&15; element e -> K = kb*32 + kpart + e
            v16h b = ldsLoadFrag16(&sF[p][kb * 32 + kpart]);
            c = __builtin_amdgcn_wmma_f32_16x16x32_f16(false, a0[kb], false, b,
                                                       (short)0, c, false, false);
        }
        v8h hv;
#pragma unroll
        for (int r = 0; r < 8; ++r) {
            float v = c[r] > 0.0f ? c[r] : 0.0f;
            hv[r] = (_Float16)v;
        }
        *(v8h*)&sH[p][rbase] = hv;              // one packed ds_store_b128
    }

    __syncthreads();

    // ------------- layer 1 (+ fused layer 2 dot with w2) ---------------
    float w2v[8];
#pragma unroll
    for (int r = 0; r < 8; ++r) w2v[r] = sW2[rbase + r];

    for (int n = 0; n < 8; ++n) {
        const int p = n * 16 + col;
        v8f c;
#pragma unroll
        for (int r = 0; r < 8; ++r) c[r] = sB1[rbase + r];
#pragma unroll
        for (int kb = 0; kb < 4; ++kb) {
            v16h b = ldsLoadFrag16(&sH[p][kb * 32 + kpart]);
            c = __builtin_amdgcn_wmma_f32_16x16x32_f16(false, a1[kb], false, b,
                                                       (short)0, c, false, false);
        }
        // h1 = relu(c); out_partial[p] += sum_r w2[row_r] * h1_r   (ds_add_f32)
        float part = 0.0f;
#pragma unroll
        for (int r = 0; r < 8; ++r) {
            float v = c[r] > 0.0f ? c[r] : 0.0f;
            part += w2v[r] * v;
        }
        atomicAdd(&sOut[p], part);
    }

    __syncthreads();

    if (t < TILE) out[(size_t)inst * NPIX + px0 + t] = sOut[t];
}

extern "C" void kernel_launch(void* const* d_in, const int* in_sizes, int n_in,
                              void* d_out, int out_size, void* d_ws, size_t ws_size,
                              hipStream_t stream) {
    const float* x       = (const float*)d_in[0];   // (4, 64, 60, 160) f32
    const float* params  = (const float*)d_in[1];   // (128, 25217) f32
    const int*   num_ins = (const int*)d_in[2];     // (4,) i32
    float* out = (float*)d_out;                     // (1, 128, 60, 160) f32

    const int n_images = in_sizes[2];
    const int n_inst   = in_sizes[1] / NPAR;        // 128

    dim3 grid(NTILES, n_inst);                      // 75 x 128 workgroups
    dyn_mask_head_kernel<<<grid, 256, 0, stream>>>(x, params, num_ins, n_images, out);
}
